// CausalSelfAttention_1194000908715
// MI455X (gfx1250) — matrix operand order repaired and checked by reading, not compiled, not run on hardware
//
#include <hip/hip_runtime.h>
#include <hip/hip_bf16.h>

// ---------------------------------------------------------------------------
// Types
// ---------------------------------------------------------------------------
typedef __bf16 bf16_t;
typedef __attribute__((ext_vector_type(16))) __bf16 v16bf;
typedef __attribute__((ext_vector_type(16))) float  v16f;
typedef __attribute__((ext_vector_type(8)))  float  v8f;
typedef __attribute__((ext_vector_type(4)))  int    v4i;

#define BM 128
#define BN 128
#define BK 32
#define LDS_STRIDE 40   // 32 + 8 halfs pad (80B row stride, keeps 16B alignment)

// ---------------------------------------------------------------------------
// CDNA5 async global->LDS path (guarded: falls back cleanly if the
// toolchain does not declare the builtins). Parameter types per hipcc
// diagnostic: (v4i AS1*, v4i AS3*, imm offset, imm cpol).
// ---------------------------------------------------------------------------
#if defined(__has_builtin)
#if __has_builtin(__builtin_amdgcn_global_load_async_to_lds_b128) && \
    __has_builtin(__builtin_amdgcn_s_wait_asynccnt)
#define ASYNC_LDS_AVAILABLE 1
#endif
#endif
#ifndef ASYNC_LDS_AVAILABLE
#define ASYNC_LDS_AVAILABLE 0
#endif

#if ASYNC_LDS_AVAILABLE
typedef __attribute__((address_space(1))) v4i* async_gptr_t;
typedef __attribute__((address_space(3))) v4i* async_lptr_t;
#endif

// ---------------------------------------------------------------------------
// Stage a 128x32 tile (rows x K) into LDS as bf16.
// 256 threads: thread t -> row = t>>1, 16 contiguous elements at col (t&1)*16.
// f32 sources are converted to bf16 on the fly; bf16 sources use the async
// global->LDS DMA path when available (32B per thread via offset:0/offset:16).
// ---------------------------------------------------------------------------
template <typename T>
__device__ __forceinline__ void stage_tile(bf16_t* __restrict__ lds,
                                           const T* __restrict__ src0,
                                           long ld, int tid) {
  const int row  = tid >> 1;
  const int col0 = (tid & 1) << 4;
  const T* src = src0 + (long)row * ld + col0;
  bf16_t* dst = lds + row * LDS_STRIDE + col0;
  if constexpr (sizeof(T) == 4) {
    union { bf16_t h[16]; uint4 u[2]; } t;
    v16f x = *reinterpret_cast<const v16f*>(src);
#pragma unroll
    for (int i = 0; i < 16; ++i) t.h[i] = (bf16_t)x[i];
    uint4* d = reinterpret_cast<uint4*>(dst);
    d[0] = t.u[0];
    d[1] = t.u[1];
  } else {
#if ASYNC_LDS_AVAILABLE
    async_gptr_t g = (async_gptr_t)(unsigned long long)src;
    async_lptr_t l = (async_lptr_t)(unsigned int)(unsigned long long)dst;
    __builtin_amdgcn_global_load_async_to_lds_b128(g, l, 0, 0);
    __builtin_amdgcn_global_load_async_to_lds_b128(g, l, 16, 0);
#else
    union { bf16_t h[16]; uint4 u[2]; } t;
    v16bf x = *reinterpret_cast<const v16bf*>(src);
#pragma unroll
    for (int i = 0; i < 16; ++i) t.h[i] = x[i];
    uint4* d = reinterpret_cast<uint4*>(dst);
    d[0] = t.u[0];
    d[1] = t.u[1];
#endif
  }
}

// ---------------------------------------------------------------------------
// Generic NT GEMM:  C[M,N] = scale * (A[M,K] @ B[N,K]^T) (+ bias[n]) (+ causal)
// Both A and B are row-major with K contiguous. Output f32 or bf16.
// Block: 256 threads = 8 waves; tile 128x128x32; wave -> 32x64 (2x4 WMMA tiles).
// ---------------------------------------------------------------------------
template <typename TA, typename TB, bool CAUSAL, bool OUT_BF16, bool HAS_BIAS>
__global__ __launch_bounds__(256) void gemm_nt_kernel(
    const TA* __restrict__ A, long strideAz,
    const TB* __restrict__ Bm, long strideBz,
    const float* __restrict__ bias,
    void* __restrict__ Cout, long strideCz,
    int M, int N, int K, float scale) {
  __shared__ bf16_t sA[BM * LDS_STRIDE];
  __shared__ bf16_t sB[BN * LDS_STRIDE];

  const int tid  = threadIdx.x;
  const int lane = tid & 31;
  const int wave = tid >> 5;
  const int wm   = wave & 3;   // 4 waves along M (32 rows each)
  const int wn   = wave >> 2;  // 2 waves along N (64 cols each)

  const int m0 = blockIdx.y * BM;
  const int n0 = blockIdx.x * BN;
  const int z  = blockIdx.z;

  const TA* Ab = A  + (long)z * strideAz;
  const TB* Bb = Bm + (long)z * strideBz;

  if constexpr (CAUSAL) {
    // Tile entirely above the diagonal: write mask value, skip all math.
    if (n0 > m0 + (BM - 1)) {
      if constexpr (OUT_BF16) {
        bf16_t* Cb = (bf16_t*)Cout + (long)z * strideCz;
        for (int i = tid; i < BM * BN; i += 256)
          Cb[(long)(m0 + (i >> 7)) * N + (n0 + (i & (BN - 1)))] = (bf16_t)(-3.0e38f);
      } else {
        float* Cf = (float*)Cout + (long)z * strideCz;
        for (int i = tid; i < BM * BN; i += 256)
          Cf[(long)(m0 + (i >> 7)) * N + (n0 + (i & (BN - 1)))] = -3.0e38f;
      }
      return;
    }
  }

  v8f acc[2][4] = {};
  const int half = lane >> 4;   // K-half select per ISA 16-bit layouts
  const int l16  = lane & 15;

  for (int k0 = 0; k0 < K; k0 += BK) {
    stage_tile(sA, Ab + (long)m0 * K + k0, K, tid);
    stage_tile(sB, Bb + (long)n0 * K + k0, K, tid);
    if (k0 + BK < K) {
      __builtin_prefetch(Ab + (long)(m0 + (tid >> 1)) * K + k0 + BK, 0, 0);
      __builtin_prefetch(Bb + (long)(n0 + (tid >> 1)) * K + k0 + BK, 0, 0);
    }
#if ASYNC_LDS_AVAILABLE
    if constexpr (sizeof(TA) == 2 || sizeof(TB) == 2)
      __builtin_amdgcn_s_wait_asynccnt(0);
#endif
    __syncthreads();

    union Frag { uint4 u[2]; v16bf v; };
    Frag a[2], b[4];
#pragma unroll
    for (int r = 0; r < 2; ++r) {
      // A 16x32 bf16: lanes 0-15 hold K={0..7,16..23}, lanes 16-31 K={8..15,24..31}
      const bf16_t* rowp = sA + (wm * 32 + r * 16 + l16) * LDS_STRIDE;
      a[r].u[0] = *reinterpret_cast<const uint4*>(rowp + half * 8);
      a[r].u[1] = *reinterpret_cast<const uint4*>(rowp + 16 + half * 8);
    }
#pragma unroll
    for (int c = 0; c < 4; ++c) {
      // B 32x16 bf16: lanes 0-15 hold K={0..7,16..23}, lanes 16-31 K={8..15,24..31}
      const bf16_t* rowp = sB + (wn * 64 + c * 16 + l16) * LDS_STRIDE;
      b[c].u[0] = *reinterpret_cast<const uint4*>(rowp + half * 8);
      b[c].u[1] = *reinterpret_cast<const uint4*>(rowp + 16 + half * 8);
    }
#pragma unroll
    for (int r = 0; r < 2; ++r)
#pragma unroll
      for (int c = 0; c < 4; ++c)
        acc[r][c] = __builtin_amdgcn_wmma_f32_16x16x32_bf16(
            false, a[r].v, false, b[c].v, (short)0, acc[r][c], false, false);
    __syncthreads();
  }

  // Epilogue. C/D 16x16 f32 layout: lane = N within 16, VGPR v -> M = v + 8*(lane>=16).
  float*  Cf = nullptr;
  bf16_t* Cb = nullptr;
  if constexpr (OUT_BF16) Cb = (bf16_t*)Cout + (long)z * strideCz;
  else                    Cf = (float*)Cout + (long)z * strideCz;
  const int mofs = half * 8;
#pragma unroll
  for (int r = 0; r < 2; ++r) {
    const int mbase = m0 + wm * 32 + r * 16 + mofs;
#pragma unroll
    for (int c = 0; c < 4; ++c) {
      const int n = n0 + wn * 64 + c * 16 + l16;
      const float bval = HAS_BIAS ? bias[n] : 0.0f;
#pragma unroll
      for (int v = 0; v < 8; ++v) {
        const int m = mbase + v;
        float val = acc[r][c][v] * scale + bval;
        if constexpr (CAUSAL) { if (n > m) val = -3.0e38f; }
        if constexpr (OUT_BF16) Cb[(long)m * N + n] = (bf16_t)val;
        else                    Cf[(long)m * N + n] = val;
      }
    }
  }
}

// ---------------------------------------------------------------------------
// bf16 tile transpose: in[z][rows][cols] -> out[z][cols][rows]
// ---------------------------------------------------------------------------
__global__ __launch_bounds__(256) void transpose_bf16_kernel(
    const bf16_t* __restrict__ in, bf16_t* __restrict__ out, int rows, int cols) {
  __shared__ bf16_t tile[32][33];
  const int z = blockIdx.z;
  const bf16_t* ib = in  + (long)z * rows * cols;
  bf16_t*       ob = out + (long)z * rows * cols;
  const int c0 = blockIdx.x * 32, r0 = blockIdx.y * 32;
  const int tx = threadIdx.x & 31, ty = threadIdx.x >> 5;
#pragma unroll
  for (int i = 0; i < 32; i += 8)
    tile[ty + i][tx] = ib[(long)(r0 + ty + i) * cols + (c0 + tx)];
  __syncthreads();
#pragma unroll
  for (int i = 0; i < 32; i += 8)
    ob[(long)(c0 + ty + i) * rows + (r0 + tx)] = tile[tx][ty + i];
}

// ---------------------------------------------------------------------------
// Row softmax: S[row, 0..T) f32 -> P[row, 0..T) bf16. One 256-thread block/row.
// Masked entries are -3e38 -> exp underflows to 0 (row max is always finite).
// ---------------------------------------------------------------------------
__global__ __launch_bounds__(256) void softmax_rows_kernel(
    const float* __restrict__ S, bf16_t* __restrict__ P, int T) {
  __shared__ float red[8];
  const long row = blockIdx.x;
  const float* s = S + row * (long)T;
  bf16_t*      p = P + row * (long)T;
  const int tid = threadIdx.x, lane = tid & 31, wave = tid >> 5;

  float m = -3.0e38f;
  for (int j = tid; j < T; j += 256) m = fmaxf(m, s[j]);
#pragma unroll
  for (int off = 16; off; off >>= 1) m = fmaxf(m, __shfl_xor(m, off, 32));
  if (lane == 0) red[wave] = m;
  __syncthreads();
  float rmax = red[0];
#pragma unroll
  for (int i = 1; i < 8; ++i) rmax = fmaxf(rmax, red[i]);
  __syncthreads();

  const float LOG2E = 1.4426950408889634f;
  float sum = 0.0f;
  for (int j = tid; j < T; j += 256) sum += exp2f((s[j] - rmax) * LOG2E);
#pragma unroll
  for (int off = 16; off; off >>= 1) sum += __shfl_xor(sum, off, 32);
  if (lane == 0) red[wave] = sum;
  __syncthreads();
  float rsum = 0.0f;
#pragma unroll
  for (int i = 0; i < 8; ++i) rsum += red[i];
  const float inv = 1.0f / rsum;
  for (int j = tid; j < T; j += 256)
    p[j] = (bf16_t)(exp2f((s[j] - rmax) * LOG2E) * inv);
}

// ---------------------------------------------------------------------------
// Host-side launch
// ---------------------------------------------------------------------------
extern "C" void kernel_launch(void* const* d_in, const int* in_sizes, int n_in,
                              void* d_out, int out_size, void* d_ws, size_t ws_size,
                              hipStream_t stream) {
  (void)in_sizes; (void)n_in; (void)out_size; (void)ws_size;

  constexpr int Bb = 4, T = 2048, C = 1024;
  constexpr int M = Bb * T;                       // 8192
  constexpr size_t MB = 1024ull * 1024ull;

  const float* q   = (const float*)d_in[0];
  const float* k   = (const float*)d_in[1];
  const float* v   = (const float*)d_in[2];
  const float* Wq  = (const float*)d_in[3];
  const float* bq  = (const float*)d_in[4];
  const float* Wk  = (const float*)d_in[5];
  const float* bk  = (const float*)d_in[6];
  const float* Wv  = (const float*)d_in[7];
  const float* bv  = (const float*)d_in[8];
  const float* Wff = (const float*)d_in[9];
  const float* bff = (const float*)d_in[10];
  float* out = (float*)d_out;

  // Workspace layout (128 MB total, with lifetime-safe reuse):
  //   [0,16M)   Qp bf16   -> later P[0..16M)
  //   [16,32M)  Kp bf16   -> later P[16..32M)
  //   [32,48M)  Vp bf16   -> later O
  //   [48,64M)  VpT bf16
  //   [64,128M) S f32
  char* w = (char*)d_ws;
  bf16_t* Qp  = (bf16_t*)(w);
  bf16_t* Kp  = (bf16_t*)(w + 16 * MB);
  bf16_t* Vp  = (bf16_t*)(w + 32 * MB);
  bf16_t* P   = (bf16_t*)(w);            // reuses Qp+Kp (dead after scores)
  bf16_t* O   = (bf16_t*)(w + 32 * MB);  // reuses Vp (dead after transpose)
  bf16_t* VpT = (bf16_t*)(w + 48 * MB);
  float*  S   = (float*)(w + 64 * MB);

  const dim3 blk(256);
  const float inv_sqrt_d = 0.03125f;  // 1/sqrt(1024)

  // 1) QKV projections: [8192,1024] = f32 @ f32^T + bias -> bf16
  gemm_nt_kernel<float, float, false, true, true>
      <<<dim3(C / BN, M / BM, 1), blk, 0, stream>>>(q, 0, Wq, 0, bq, Qp, 0, M, C, C, 1.0f);
  gemm_nt_kernel<float, float, false, true, true>
      <<<dim3(C / BN, M / BM, 1), blk, 0, stream>>>(k, 0, Wk, 0, bk, Kp, 0, M, C, C, 1.0f);
  gemm_nt_kernel<float, float, false, true, true>
      <<<dim3(C / BN, M / BM, 1), blk, 0, stream>>>(v, 0, Wv, 0, bv, Vp, 0, M, C, C, 1.0f);

  // 2) Transpose V per batch: [T,C] -> [C,T] so PV becomes an NT GEMM
  transpose_bf16_kernel<<<dim3(C / 32, T / 32, Bb), blk, 0, stream>>>(Vp, VpT, T, C);

  // 3) Causal scores: S = (Q @ K^T) / sqrt(D), masked -> f32
  gemm_nt_kernel<bf16_t, bf16_t, true, false, false>
      <<<dim3(T / BN, T / BM, Bb), blk, 0, stream>>>(
          Qp, (long)T * C, Kp, (long)T * C, nullptr, S, (long)T * T, T, T, C, inv_sqrt_d);

  // 4) Row softmax -> bf16 P
  softmax_rows_kernel<<<dim3(Bb * T), blk, 0, stream>>>(S, P, T);

  // 5) O = P @ V  (V transposed: NT GEMM) -> bf16
  gemm_nt_kernel<bf16_t, bf16_t, false, true, false>
      <<<dim3(C / BN, T / BM, Bb), blk, 0, stream>>>(
          P, (long)T * T, VpT, (long)C * T, nullptr, O, (long)T * C, T, C, T, 1.0f);

  // 6) Final projection: out = O @ Wff^T + bff -> f32
  gemm_nt_kernel<bf16_t, float, false, false, true>
      <<<dim3(C / BN, M / BM, 1), blk, 0, stream>>>(O, 0, Wff, 0, bff, out, 0, M, C, C, 1.0f);
}
